// MultiheadAttention_17892833755629
// MI455X (gfx1250) — compile-verified, matrix operand order checked by reading, not run on hardware
//
#include <hip/hip_runtime.h>
#include <hip/hip_bf16.h>

// ---------------------------------------------------------------------------
// MI455X (gfx1250) multi-head attention: bf16 WMMA GEMMs + flash attention
// with TDM (tensor_load_to_lds) K-tile staging and DPP8 softmax reductions.
// S=1024, B=8, H=768, heads=12, d=64. wave32.
// ---------------------------------------------------------------------------

typedef __bf16 bf16_t;
typedef __attribute__((ext_vector_type(16))) __bf16 v16bf;
typedef __attribute__((ext_vector_type(8)))  float  v8f;
typedef __attribute__((ext_vector_type(8)))  unsigned int v8u;
typedef __attribute__((ext_vector_type(4)))  unsigned int u32x4;
typedef __attribute__((ext_vector_type(8)))  int i32x8;
typedef __attribute__((ext_vector_type(4)))  int i32x4;

#define SEQ   1024
#define BATCH 8
#define HID   768
#define NH    12
#define DH    64
#define NHEADS_TOT (BATCH * NH)            // 96
#define QKV_PLANE  (NHEADS_TOT * SEQ * DH) // 6291456 bf16 elems per q/k/v

// ---- bf16 helpers (round-to-nearest-even) ---------------------------------
__device__ __forceinline__ unsigned short bf16_bits(float x) {
  unsigned u = __builtin_bit_cast(unsigned, x);
  u += 0x7FFFu + ((u >> 16) & 1u);
  return (unsigned short)(u >> 16);
}
__device__ __forceinline__ unsigned pack2_bf16(float a, float b) {
  unsigned ua = __builtin_bit_cast(unsigned, a);
  unsigned ub = __builtin_bit_cast(unsigned, b);
  ua += 0x7FFFu + ((ua >> 16) & 1u);
  ub += 0x7FFFu + ((ub >> 16) & 1u);
  return (ua >> 16) | (ub & 0xFFFF0000u);
}

// Build a 16-bit-element WMMA A/B fragment from a row base pointer.
// ISA 16-bit 16x32 layout: element pair i covers K = 2*(i&3) + 8*half + 16*(i>>2).
__device__ __forceinline__ v16bf frag16(const unsigned short* rowbase, int half) {
  v8u u;
#pragma unroll
  for (int i = 0; i < 8; ++i) {
    int k = 2 * (i & 3) + 8 * half + 16 * (i >> 2);
    u[i] = *(const unsigned int*)(rowbase + k);
  }
  return __builtin_bit_cast(v16bf, u);
}

// ---- wave32 16-lane reductions: DPP8 for xor1/2/4 (VALU, co-executes with
// WMMA), ds_swizzle (no LDS memory) for xor8. ---------------------------------
#define DPP8SEL(a,b,c,d,e,f,g,h) \
  ((a)|((b)<<3)|((c)<<6)|((d)<<9)|((e)<<12)|((f)<<15)|((g)<<18)|((h)<<21))
#define DPP8F(v, sel) \
  __builtin_bit_cast(float, __builtin_amdgcn_mov_dpp8(__builtin_bit_cast(int, (v)), (sel)))

__device__ __forceinline__ float rmax16(float v) {
  v = fmaxf(v, DPP8F(v, DPP8SEL(1,0,3,2,5,4,7,6)));   // xor 1
  v = fmaxf(v, DPP8F(v, DPP8SEL(2,3,0,1,6,7,4,5)));   // xor 2
  v = fmaxf(v, DPP8F(v, DPP8SEL(4,5,6,7,0,1,2,3)));   // xor 4
  int t = __builtin_amdgcn_ds_swizzle(__builtin_bit_cast(int, v), 0x201F); // xor 8
  return fmaxf(v, __builtin_bit_cast(float, t));
}
__device__ __forceinline__ float rsum16(float v) {
  v += DPP8F(v, DPP8SEL(1,0,3,2,5,4,7,6));
  v += DPP8F(v, DPP8SEL(2,3,0,1,6,7,4,5));
  v += DPP8F(v, DPP8SEL(4,5,6,7,0,1,2,3));
  int t = __builtin_amdgcn_ds_swizzle(__builtin_bit_cast(int, v), 0x201F);
  return v + __builtin_bit_cast(float, t);
}

// ---------------------------------------------------------------------------
// GEMM: C[M,N] = A[M,K] * W[N,K]^T + bias.  MODE 0: plain f32 store.
// MODE 1: QKV scatter (bf16, head-major, q scaled by 0.125).
// Block tile 64x128, 256 threads = 8 waves (2 row x 4 col groups of 32x32),
// 4 WMMAs per wave per K-step (A/B fragment reuse), K step 32.
// ---------------------------------------------------------------------------
#define TM 64
#define TN 128
#define TKK 32
#define LDP 34   // padded LDS stride (bf16 elems): conflict-free, 4B aligned

__device__ __forceinline__ void store_qkv_elem(unsigned short* qkv, int m, int n, float v) {
  int which = n / HID;           // 0=q 1=k 2=v
  int r = n - which * HID;
  int head = r >> 6, dcol = r & 63;
  if (which == 0) v *= 0.125f;   // QUERY_SCALE
  int s = m >> 3, b = m & 7;     // row m = s*B + b, B = 8
  int hidx = b * NH + head;      // (B*h) index
  size_t off = (size_t)which * QKV_PLANE + ((size_t)hidx * SEQ + s) * DH + dcol;
  qkv[off] = bf16_bits(v);
}

template <int MODE>
__global__ __launch_bounds__(256) void gemm_bf16_wmma(
    const float* __restrict__ A, const float* __restrict__ W,
    const float* __restrict__ bias, float* __restrict__ C,
    unsigned short* __restrict__ qkv, int M, int N, int K) {
  __shared__ unsigned short As[TM * LDP];
  __shared__ unsigned short Bs[TN * LDP];

  const int tid  = threadIdx.x;
  const int wave = tid >> 5, lane = tid & 31;
  const int half = lane >> 4, l15 = lane & 15;
  const int wr = wave & 1;   // 32-row group
  const int wc = wave >> 1;  // 32-col group
  const int rowBase = blockIdx.x * TM;
  const int colBase = blockIdx.y * TN;

  v8f acc00 = 0.0f, acc01 = 0.0f, acc10 = 0.0f, acc11 = 0.0f;

  const int ldr = tid >> 3;        // 0..31
  const int ldk = (tid & 7) * 4;   // 0..28 step 4

  for (int k0 = 0; k0 < K; k0 += TKK) {
#pragma unroll
    for (int rr = 0; rr < 2; ++rr) {   // A: 64 rows
      const int r = ldr + rr * 32;
      const float4 a4 = *(const float4*)(A + (size_t)(rowBase + r) * K + k0 + ldk);
      *(unsigned int*)(As + r * LDP + ldk)     = pack2_bf16(a4.x, a4.y);
      *(unsigned int*)(As + r * LDP + ldk + 2) = pack2_bf16(a4.z, a4.w);
    }
#pragma unroll
    for (int rr = 0; rr < 4; ++rr) {   // B: 128 rows
      const int r = ldr + rr * 32;
      const float4 b4 = *(const float4*)(W + (size_t)(colBase + r) * K + k0 + ldk);
      *(unsigned int*)(Bs + r * LDP + ldk)     = pack2_bf16(b4.x, b4.y);
      *(unsigned int*)(Bs + r * LDP + ldk + 2) = pack2_bf16(b4.z, b4.w);
    }
    __syncthreads();

    if (k0 + TKK < K) {  // gfx1250 global_prefetch_b8 of next tiles
      __builtin_prefetch(A + (size_t)(rowBase + ldr) * K + k0 + TKK + ldk, 0, 1);
      __builtin_prefetch(W + (size_t)(colBase + ldr) * K + k0 + TKK + ldk, 0, 1);
    }

    const v16bf aF0 = frag16(As + (wr * 32 + l15) * LDP, half);
    const v16bf aF1 = frag16(As + (wr * 32 + 16 + l15) * LDP, half);
    const v16bf bF0 = frag16(Bs + (wc * 32 + l15) * LDP, half);
    const v16bf bF1 = frag16(Bs + (wc * 32 + 16 + l15) * LDP, half);
    acc00 = __builtin_amdgcn_wmma_f32_16x16x32_bf16(false, aF0, false, bF0, (short)0, acc00, false, false);
    acc01 = __builtin_amdgcn_wmma_f32_16x16x32_bf16(false, aF0, false, bF1, (short)0, acc01, false, false);
    acc10 = __builtin_amdgcn_wmma_f32_16x16x32_bf16(false, aF1, false, bF0, (short)0, acc10, false, false);
    acc11 = __builtin_amdgcn_wmma_f32_16x16x32_bf16(false, aF1, false, bF1, (short)0, acc11, false, false);
    __syncthreads();
  }

  // Epilogue. D layout: VGPR j, lane l -> M = j + 8*half, N = l15.
#pragma unroll
  for (int j = 0; j < 8; ++j) {
    const int m0 = rowBase + wr * 32 + j + 8 * half;
    const int m1 = m0 + 16;
    const int n0 = colBase + wc * 32 + l15;
    const int n1 = n0 + 16;
    const float v00 = acc00[j] + bias[n0];
    const float v01 = acc01[j] + bias[n1];
    const float v10 = acc10[j] + bias[n0];
    const float v11 = acc11[j] + bias[n1];
    if (MODE == 0) {
      C[(size_t)m0 * N + n0] = v00;
      C[(size_t)m0 * N + n1] = v01;
      C[(size_t)m1 * N + n0] = v10;
      C[(size_t)m1 * N + n1] = v11;
    } else {
      store_qkv_elem(qkv, m0, n0, v00);
      store_qkv_elem(qkv, m0, n1, v01);
      store_qkv_elem(qkv, m1, n0, v10);
      store_qkv_elem(qkv, m1, n1, v11);
    }
  }
}

// ---------------------------------------------------------------------------
// Flash attention: one block = (head n, 64-query tile), 128 threads (4 waves,
// 16 query rows each). K tile staged via the Tensor Data Mover (with HW
// padding reproducing the KLD=72 layout); V tile transposed manually.
// ---------------------------------------------------------------------------
#define KT 32
#define KLD 72   // 32 dwords data + 4 dwords TDM pad per 64-elem row
#define VLD 34   // transposed V tile stride
#define PLD 34   // per-wave probs patch stride
#define LOG2E 1.4426950408889634f

__global__ __launch_bounds__(128) void attn_flash(
    const unsigned short* __restrict__ q, const unsigned short* __restrict__ k,
    const unsigned short* __restrict__ v, float* __restrict__ ctx) {
  __shared__ unsigned short Ks[KT * KLD];       // [key][d], TDM-padded rows
  __shared__ unsigned short Vt[DH * VLD];       // [d][key]  (transposed)
  __shared__ unsigned short Ps[4 * 16 * PLD];   // per-wave probs [16][32]

  const int n   = blockIdx.x;      // 0..95 (b*12 + head)
  const int qt  = blockIdx.y;      // 0..15
  const int tid = threadIdx.x;
  const int wave = tid >> 5, lane = tid & 31;
  const int half = lane >> 4, l15 = lane & 15;
  const int qrow0 = qt * 64 + wave * 16;

  const unsigned short* qb = q + (size_t)n * SEQ * DH;
  const unsigned short* kb = k + (size_t)n * SEQ * DH;
  const unsigned short* vb = v + (size_t)n * SEQ * DH;

  // Q fragments (A-matrix, M = l15): d-chunks [0,32) and [32,64), pinned in VGPRs.
  v8u qa0, qa1;
  {
    const unsigned short* qr = qb + (size_t)(qrow0 + l15) * DH;
#pragma unroll
    for (int i = 0; i < 8; ++i) {
      int kk = 2 * (i & 3) + 8 * half + 16 * (i >> 2);
      qa0[i] = *(const unsigned int*)(qr + kk);
      qa1[i] = *(const unsigned int*)(qr + 32 + kk);
    }
  }
  const v16bf qF0 = __builtin_bit_cast(v16bf, qa0);
  const v16bf qF1 = __builtin_bit_cast(v16bf, qa1);

  v8f c0 = 0.0f, c1 = 0.0f, c2 = 0.0f, c3 = 0.0f;
  float mrow[8], lsum[8];
#pragma unroll
  for (int j = 0; j < 8; ++j) { mrow[j] = -__builtin_inff(); lsum[j] = 0.0f; }

  unsigned short* Pw = Ps + wave * 16 * PLD;
  const unsigned ks_lds = (unsigned)(uintptr_t)(&Ks[0]);  // LDS byte offset (addr[31:0])

  for (int kt0 = 0; kt0 < SEQ; kt0 += KT) {
    // --- K tile via TDM: 32x64 bf16 2D tile, pad 4 dwords every 32 dwords
    //     so LDS rows land on the KLD=72 stride. Issued by wave 0 only.
    if (tid < 32) {
      const unsigned long long ga = (unsigned long long)(uintptr_t)(kb + (size_t)kt0 * DH);
      u32x4 g0;
      g0[0] = 1u;                                     // count=1, user mode
      g0[1] = ks_lds;                                 // lds_addr
      g0[2] = (unsigned)(ga & 0xFFFFFFFFu);           // global_addr[31:0]
      g0[3] = (unsigned)((ga >> 32) & 0x01FFFFFFu) | (2u << 30);  // addr[56:32] | type=2
      i32x8 g1;
      g1[0] = (1 << 16) | (1 << 20) | (4 << 22) | (3 << 25); // data_size=2B, pad_en, ivl=32dw, amt=4dw
      g1[1] = (int)(64u << 16);                       // tensor_dim0 = 64 (d)
      g1[2] = (int)((unsigned)SEQ << 16);             // tensor_dim1 = 1024 (keys)
      g1[3] = (int)(64u << 16);                       // tile_dim0 = 64
      g1[4] = KT;                                     // tile_dim1 = 32
      g1[5] = 64;                                     // tensor_dim0_stride = 64
      g1[6] = 0; g1[7] = 0;
      const i32x4 gz4 = {0, 0, 0, 0};                 // 2D: groups 2/3 unused
      const i32x8 gz8 = {0, 0, 0, 0, 0, 0, 0, 0};
      __builtin_amdgcn_tensor_load_to_lds(g0, g1, gz4, gz4, gz8, 0);
    }
    // --- V tile: cooperative transposed load (TDM cannot transpose).
    {
      const int krow = tid >> 2;          // 0..31
      const int db   = (tid & 3) * 16;    // 0,16,32,48
      const uint4 vx0 = *(const uint4*)(vb + (size_t)(kt0 + krow) * DH + db);
      const uint4 vx1 = *(const uint4*)(vb + (size_t)(kt0 + krow) * DH + db + 8);
      const unsigned int w[8] = {vx0.x, vx0.y, vx0.z, vx0.w, vx1.x, vx1.y, vx1.z, vx1.w};
#pragma unroll
      for (int i = 0; i < 8; ++i) {       // scatter into Vt[d][key]
        Vt[(db + 2 * i)     * VLD + krow] = (unsigned short)w[i];
        Vt[(db + 2 * i + 1) * VLD + krow] = (unsigned short)(w[i] >> 16);
      }
    }
    if (tid < 32) __builtin_amdgcn_s_wait_tensorcnt(0);
    __syncthreads();

    // scores = Q(16x64) * K^T(64x32): two 16x16 D tiles, 2 d-chunks each.
    v8f s0 = 0.0f, s1 = 0.0f;
    s0 = __builtin_amdgcn_wmma_f32_16x16x32_bf16(false, qF0, false, frag16(Ks + l15 * KLD, half),        (short)0, s0, false, false);
    s0 = __builtin_amdgcn_wmma_f32_16x16x32_bf16(false, qF1, false, frag16(Ks + l15 * KLD + 32, half),   (short)0, s0, false, false);
    s1 = __builtin_amdgcn_wmma_f32_16x16x32_bf16(false, qF0, false, frag16(Ks + (16 + l15) * KLD, half),      (short)0, s1, false, false);
    s1 = __builtin_amdgcn_wmma_f32_16x16x32_bf16(false, qF1, false, frag16(Ks + (16 + l15) * KLD + 32, half), (short)0, s1, false, false);

    // Online softmax. Row r = j + 8*half lives across the 16 lanes of `half`.
#pragma unroll
    for (int j = 0; j < 8; ++j) {
      const float rm    = rmax16(fmaxf(s0[j], s1[j]));
      const float mnew  = fmaxf(mrow[j], rm);
      const float scale = exp2f((mrow[j] - mnew) * LOG2E);
      const float p0 = exp2f((s0[j] - mnew) * LOG2E);
      const float p1 = exp2f((s1[j] - mnew) * LOG2E);
      const float ps = rsum16(p0 + p1);
      lsum[j] = lsum[j] * scale + ps;
      mrow[j] = mnew;
      c0[j] *= scale; c1[j] *= scale; c2[j] *= scale; c3[j] *= scale;
      // stash probs (D layout) into per-wave LDS patch as bf16
      Pw[(j + 8 * half) * PLD + l15]      = bf16_bits(p0);
      Pw[(j + 8 * half) * PLD + 16 + l15] = bf16_bits(p1);
    }

    // Re-read probs in A layout (same-wave LDS, in-order: no barrier needed).
    const v16bf pF = frag16(Pw + l15 * PLD, half);

    // ctx += P(16x32) * V(32x64): 4 d-column tiles.
    c0 = __builtin_amdgcn_wmma_f32_16x16x32_bf16(false, pF, false, frag16(Vt + l15 * VLD, half),        (short)0, c0, false, false);
    c1 = __builtin_amdgcn_wmma_f32_16x16x32_bf16(false, pF, false, frag16(Vt + (16 + l15) * VLD, half), (short)0, c1, false, false);
    c2 = __builtin_amdgcn_wmma_f32_16x16x32_bf16(false, pF, false, frag16(Vt + (32 + l15) * VLD, half), (short)0, c2, false, false);
    c3 = __builtin_amdgcn_wmma_f32_16x16x32_bf16(false, pF, false, frag16(Vt + (48 + l15) * VLD, half), (short)0, c3, false, false);
    __syncthreads();
  }

  // Normalize and store ctx in (s, b, H) f32 layout for the out-projection.
  const int b = n / NH, head = n - b * NH;
#pragma unroll
  for (int j = 0; j < 8; ++j) {
    const float inv = 1.0f / lsum[j];
    const int s = qrow0 + j + 8 * half;
    float* dst = ctx + ((size_t)(s * BATCH + b)) * HID + head * DH + l15;
    dst[0]  = c0[j] * inv;
    dst[16] = c1[j] * inv;
    dst[32] = c2[j] * inv;
    dst[48] = c3[j] * inv;
  }
}

// ---------------------------------------------------------------------------
extern "C" void kernel_launch(void* const* d_in, const int* in_sizes, int n_in,
                              void* d_out, int out_size, void* d_ws, size_t ws_size,
                              hipStream_t stream) {
  const float* X  = (const float*)d_in[0];  // (1024, 8, 768)
  const float* Wi = (const float*)d_in[1];  // (2304, 768)
  const float* bi = (const float*)d_in[2];  // (2304,)
  const float* Wo = (const float*)d_in[3];  // (768, 768)
  const float* bo = (const float*)d_in[4];  // (768,)
  float* out = (float*)d_out;               // (1024, 8, 768)

  unsigned short* qkv = (unsigned short*)d_ws;            // 3 * 6291456 bf16
  float* ctx = (float*)((char*)d_ws + (size_t)3 * QKV_PLANE * sizeof(unsigned short));

  const int M = SEQ * BATCH;  // 8192

  // 1) fused QKV projection -> bf16 head-major q/k/v (q pre-scaled)
  gemm_bf16_wmma<1><<<dim3(M / TM, (3 * HID) / TN), 256, 0, stream>>>(
      X, Wi, bi, nullptr, qkv, M, 3 * HID, HID);

  // 2) flash attention per (head, 64-query tile)
  attn_flash<<<dim3(NHEADS_TOT, SEQ / 64), 128, 0, stream>>>(
      qkv, qkv + QKV_PLANE, qkv + 2 * (size_t)QKV_PLANE, ctx);

  // 3) output projection -> f32 result
  gemm_bf16_wmma<0><<<dim3(M / TM, HID / TN), 256, 0, stream>>>(
      ctx, Wo, bo, out, nullptr, M, HID, HID);
}
